// Qwen3Attention_37598143709441
// MI455X (gfx1250) — matrix operand order repaired and checked by reading, not compiled, not run on hardware
//
#include <hip/hip_runtime.h>
#include <hip/hip_bf16.h>

#define T_SEQ 2048
#define HIDN  2048
#define NH    16
#define NKV   8
#define HD    128
#define QKV_N (NH*HD + 2*NKV*HD)   // 4096
#define EPSF  1e-6f

typedef __attribute__((ext_vector_type(16))) __bf16 v16bf;
typedef __attribute__((ext_vector_type(8)))  __bf16 v8bf;
typedef __attribute__((ext_vector_type(8)))  float  v8f;
typedef __attribute__((ext_vector_type(4)))  unsigned int v4u;
typedef __attribute__((ext_vector_type(8)))  int v8i;
typedef __attribute__((ext_vector_type(4)))  int v4i;

#if __has_builtin(__builtin_amdgcn_tensor_load_to_lds)
#define HAVE_TDM 1
#else
#define HAVE_TDM 0
#endif

// ---- WMMA fragment helpers (contiguous k order, used alike for A and B here) -------------------
// A-matrix 16x32 bf16: lanes 0-15 row=lane, elements = K{0..15};
// lanes 16-31 same rows, K{16..31}. Two contiguous 8-element chunks.
__device__ __forceinline__ v16bf load_a_frag(const __bf16* row, int k0, bool hi) {
  const int sh = hi ? 16 : 0;
  v8bf lo = *(const v8bf*)(row + k0 + sh);
  v8bf hh = *(const v8bf*)(row + k0 + 8 + sh);
  v16bf a;
#pragma unroll
  for (int i = 0; i < 8; ++i) { a[i] = lo[i]; a[i + 8] = hh[i]; }
  return a;
}

__device__ __forceinline__ v8f wmma_bf16(v16bf a, v16bf b, v8f c) {
  return __builtin_amdgcn_wmma_f32_16x16x32_bf16(false, a, false, b,
                                                 (short)0, c, false, false);
}

// LDS byte offset of a pointer into a __shared__ array.
typedef __attribute__((address_space(3))) const void lds_cv;
__device__ __forceinline__ unsigned lds_offset(const void* p) {
  return (unsigned)(__SIZE_TYPE__)(lds_cv*)p;
}

#if HAVE_TDM
// Issue one 2D TDM tile load: tile1 rows x tile0 bf16 elements, row stride
// stride0 elements, packed row-major into LDS at lds_off. D# per ISA ch.8.
__device__ __forceinline__ void tdm_load_2d(unsigned lds_off, const __bf16* gaddr,
                                            unsigned td0, unsigned td1,
                                            unsigned tile0, unsigned tile1,
                                            unsigned stride0) {
  unsigned long long ga = (unsigned long long)(__SIZE_TYPE__)gaddr;
  v4u g0;
  g0[0] = 1u;                                    // count=1, user descriptor
  g0[1] = lds_off;                               // lds_addr
  g0[2] = (unsigned)(ga & 0xffffffffu);          // global_addr[31:0]
  g0[3] = (unsigned)((ga >> 32) & 0x01ffffffu)   // global_addr[56:32]
        | (2u << 30);                            // type=2 ("image")
  v8i g1;
  g1[0] = (int)(1u << 16);                       // data_size=1 -> 2 bytes
  g1[1] = (int)((td0 & 0xffffu) << 16);          // tensor_dim0[15:0]
  g1[2] = (int)(((td0 >> 16) & 0xffffu) | ((td1 & 0xffffu) << 16));
  g1[3] = (int)(((td1 >> 16) & 0xffffu) | (tile0 << 16));   // tile_dim0
  g1[4] = (int)(tile1 & 0xffffu);                // tile_dim1, tile_dim2=0
  g1[5] = (int)stride0;                          // tensor_dim0_stride[31:0]
  g1[6] = 0;
  g1[7] = 0;
  v4i z = {0, 0, 0, 0};
#if defined(__clang_major__) && __clang_major__ >= 23
  v8i z8 = {0, 0, 0, 0, 0, 0, 0, 0};
  __builtin_amdgcn_tensor_load_to_lds(g0, g1, z, z, z8, 0);
#else
  __builtin_amdgcn_tensor_load_to_lds(g0, g1, z, z, 0);
#endif
}
#endif

// ---- fp32 -> bf16 conversion ------------------------------------------------
__global__ __launch_bounds__(256) void convert_f32_bf16(
    const float* __restrict__ s, __bf16* __restrict__ d, int n) {
  int i = blockIdx.x * blockDim.x + threadIdx.x;
  if (i < n) d[i] = (__bf16)s[i];
}

// ---- bf16 WMMA GEMM, TDM double-buffered through LDS ------------------------
// C[m,n] = sum_k A[m,k] * B[n,k].  Block = 8 waves, tile 128x64, K-stage 64.
#define BM 128
#define BN 64
#define BK 64
__global__ __launch_bounds__(256) void gemm_bf16_wmma(
    const __bf16* __restrict__ A, const __bf16* __restrict__ B,
    float* __restrict__ C, int M, int N, int K) {
  __shared__ __align__(32) __bf16 sA[2][BM * BK];   // 16 KB x2
  __shared__ __align__(32) __bf16 sB[2][BN * BK];   //  8 KB x2

  const int tid = threadIdx.x;
  const int lane = tid & 31;
  const int wv = tid >> 5;
  const bool hi = lane >= 16;
  const int l = lane & 15;
  const int nb = N / BN;
  const int bm0 = (int)(blockIdx.x / nb) * BM;
  const int bn0 = (int)(blockIdx.x % nb) * BN;

#if !HAVE_TDM
  const int ar = tid >> 1, ah = (tid & 1) * 32;     // A copy: row, elem offset
  const int br = tid >> 2, bq = (tid & 3) * 16;     // B copy: row, elem offset
  const __bf16* gA = A + (size_t)(bm0 + ar) * K + ah;
  const __bf16* gB = B + (size_t)(bn0 + br) * K + bq;
#endif

  auto stage = [&](int buf, int k0) {
#if HAVE_TDM
    if (wv == 0) {   // one wave launches both tile DMAs (EXEC-independent)
      tdm_load_2d(lds_offset(&sA[buf][0]), A + (size_t)bm0 * K + k0,
                  (unsigned)(K - k0), (unsigned)(M - bm0), BK, BM, (unsigned)K);
      tdm_load_2d(lds_offset(&sB[buf][0]), B + (size_t)bn0 * K + k0,
                  (unsigned)(K - k0), (unsigned)(N - bn0), BK, BN, (unsigned)K);
    }
#else
#pragma unroll
    for (int c = 0; c < 4; ++c)
      *(v8bf*)&sA[buf][ar * BK + ah + c * 8] = *(const v8bf*)(gA + k0 + c * 8);
#pragma unroll
    for (int c = 0; c < 2; ++c)
      *(v8bf*)&sB[buf][br * BK + bq + c * 8] = *(const v8bf*)(gB + k0 + c * 8);
#endif
  };
  auto publish = [&]() {
#if HAVE_TDM
    if (wv == 0) __builtin_amdgcn_s_wait_tensorcnt(0);
#endif
    __syncthreads();
  };

  v8f acc[4] = {};
  stage(0, 0);
  publish();

  const int nk = K / BK;
  for (int kt = 0; kt < nk; ++kt) {
    const int buf = kt & 1;
    if (kt + 1 < nk) stage(buf ^ 1, (kt + 1) * BK);   // DMA next stage now

    const __bf16* la = &sA[buf][(wv * 16 + l) * BK];
#pragma unroll
    for (int ks = 0; ks < BK; ks += 32) {
      // batch all fragment loads, then issue WMMAs back-to-back
      v16bf af = load_a_frag(la, ks, hi);
      v16bf bfr[4];
#pragma unroll
      for (int j = 0; j < 4; ++j)
        bfr[j] = *(const v16bf*)(&sB[buf][(j * 16 + l) * BK + ks + (hi ? 16 : 0)]);
#pragma unroll
      for (int j = 0; j < 4; ++j) acc[j] = wmma_bf16(af, bfr[j], acc[j]);
    }
    publish();   // next buffer ready for kt+1
  }

#pragma unroll
  for (int j = 0; j < 4; ++j)
#pragma unroll
    for (int g = 0; g < 8; ++g)
      C[(size_t)(bm0 + wv * 16 + g + (hi ? 8 : 0)) * N + bn0 + j * 16 + l] =
          acc[j][g];
}

// ---- fused RMSNorm + RoPE + layout/cast into bf16 Q/K/V ---------------------
__global__ __launch_bounds__(256) void norm_rope_kernel(
    const float* __restrict__ qkv, const float* __restrict__ qnw,
    const float* __restrict__ knw, const int* __restrict__ pos,
    __bf16* __restrict__ qo, __bf16* __restrict__ ko, __bf16* __restrict__ vo) {
  const int lane = threadIdx.x & 31;
  const int wid = (blockIdx.x * blockDim.x + threadIdx.x) >> 5;
  const int t = wid >> 5;
  const int r = wid & 31;
  if (t >= T_SEQ) return;
  const float* row = qkv + (size_t)t * QKV_N;

  if (r < NH + NKV) {
    const bool isq = r < NH;
    const int h = isq ? r : r - NH;
    const float* x = row + (isq ? h * HD : NH * HD + h * HD);
    const float* w = isq ? qnw : knw;
    const int i0 = lane * 2;
    float x1a = x[i0], x1b = x[i0 + 1], x2a = x[i0 + 64], x2b = x[i0 + 65];
    float ss = x1a * x1a + x1b * x1b + x2a * x2a + x2b * x2b;
#pragma unroll
    for (int off = 16; off >= 1; off >>= 1) ss += __shfl_xor(ss, off, 32);
    float rs = rsqrtf(ss * (1.0f / HD) + EPSF);
    float p = (float)pos[t];
    float n1a = x1a * rs * w[i0],      n1b = x1b * rs * w[i0 + 1];
    float n2a = x2a * rs * w[i0 + 64], n2b = x2b * rs * w[i0 + 65];
    float aa = p * __powf(10000.0f, -(float)i0 * (1.0f / 64.0f));
    float ab = p * __powf(10000.0f, -(float)(i0 + 1) * (1.0f / 64.0f));
    float ca = __cosf(aa), sa = __sinf(aa);
    float cb = __cosf(ab), sb = __sinf(ab);
    __bf16* dst = isq ? (qo + ((size_t)t * NH + h) * HD)
                      : (ko + ((size_t)t * NKV + h) * HD);
    dst[i0]      = (__bf16)(n1a * ca - n2a * sa);
    dst[i0 + 1]  = (__bf16)(n1b * cb - n2b * sb);
    dst[i0 + 64] = (__bf16)(n2a * ca + n1a * sa);
    dst[i0 + 65] = (__bf16)(n2b * cb + n1b * sb);
  } else {
    const int h = r - (NH + NKV);
    const float* x = row + NH * HD + NKV * HD + h * HD;
    __bf16* dst = vo + ((size_t)t * NKV + h) * HD;
#pragma unroll
    for (int j = 0; j < 4; ++j) dst[lane * 4 + j] = (__bf16)x[lane * 4 + j];
  }
}

// ---- lambda scalar ----------------------------------------------------------
__global__ __launch_bounds__(128) void lambda_kernel(
    const float* __restrict__ lq1, const float* __restrict__ lk1,
    const float* __restrict__ lq2, const float* __restrict__ lk2,
    const float* __restrict__ linit, float* __restrict__ lam) {
  __shared__ float s1s[128], s2s[128];
  int i = threadIdx.x;
  s1s[i] = lq1[i] * lk1[i];
  s2s[i] = lq2[i] * lk2[i];
  __syncthreads();
  for (int off = 64; off >= 1; off >>= 1) {
    if (i < off) { s1s[i] += s1s[i + off]; s2s[i] += s2s[i + off]; }
    __syncthreads();
  }
  if (i == 0) *lam = __expf(s1s[0]) - __expf(s2s[0]) + *linit;
}

// ---- differential flash attention: one wave per (head, 16-query tile) -------
__global__ __launch_bounds__(32) void diff_attn_kernel(
    const __bf16* __restrict__ q, const __bf16* __restrict__ k,
    const __bf16* __restrict__ v, const float* __restrict__ scaler,
    const float* __restrict__ lamp, __bf16* __restrict__ out) {
  __shared__ __align__(32) __bf16 lds_vT[HD * 32];   // [d][key] transposed
  __shared__ __align__(32) __bf16 lds_p1[16 * 32];
  __shared__ __align__(32) __bf16 lds_p2[16 * 32];

  const int lane = threadIdx.x;
  const bool hi = lane >= 16;
  const int l = lane & 15;
  const int h = blockIdx.x % NH;
  const int qb = blockIdx.x / NH;
  const int qbase = qb * 16;
  const int kvh = h / (NH / NKV);
  const float sc = rsqrtf((float)HD) * scaler[h];
  const __bf16* qrow = q + ((size_t)(qbase + l) * NH + h) * HD;

  v8f o1[8] = {}, o2[8] = {};
  float m1[8], l1[8], m2[8], l2[8];
#pragma unroll
  for (int g = 0; g < 8; ++g) { m1[g] = m2[g] = -1e30f; l1[g] = l2[g] = 0.f; }

  for (int kb = 0; kb < qbase + 16; kb += 32) {
    {  // stage V transposed: lane owns key kb+lane, scatters 128 d-values
      int key = kb + lane; if (key > T_SEQ - 1) key = T_SEQ - 1;
      const __bf16* vrow = v + ((size_t)key * NKV + kvh) * HD;
      __builtin_prefetch(vrow + 32 * NKV * HD, 0, 0);
#pragma unroll
      for (int c0 = 0; c0 < HD; c0 += 8) {
        v8bf chunk = *(const v8bf*)(vrow + c0);
#pragma unroll
        for (int j = 0; j < 8; ++j) lds_vT[(c0 + j) * 32 + lane] = chunk[j];
      }
    }
    __syncthreads();

    v8f s1t[2], s2t[2];
#pragma unroll
    for (int sub = 0; sub < 2; ++sub) {
      int key = kb + sub * 16 + l; if (key > T_SEQ - 1) key = T_SEQ - 1;
      const __bf16* krow = k + ((size_t)key * NKV + kvh) * HD;
      v8f s1 = {}, s2 = {};
#pragma unroll
      for (int k0 = 0; k0 < 64; k0 += 32) {
        // batch the four fragment loads, then both WMMAs
        v16bf qa1 = load_a_frag(qrow, k0, hi);
        v16bf qa2 = load_a_frag(qrow, 64 + k0, hi);
        v16bf kf1 = *(const v16bf*)(krow + k0 + (hi ? 16 : 0));
        v16bf kf2 = *(const v16bf*)(krow + 64 + k0 + (hi ? 16 : 0));
        s1 = wmma_bf16(qa1, kf1, s1);
        s2 = wmma_bf16(qa2, kf2, s2);
      }
      const int col = kb + sub * 16 + l;
#pragma unroll
      for (int g = 0; g < 8; ++g) {
        const int row = qbase + g + (hi ? 8 : 0);
        const bool ok = (col <= row);
        s1[g] = ok ? s1[g] * sc : -1e30f;
        s2[g] = ok ? s2[g] * sc : -1e30f;
      }
      s1t[sub] = s1; s2t[sub] = s2;
    }

    // online softmax, both streams; row = g + 8*hi, reduce over 16-lane group
#pragma unroll
    for (int g = 0; g < 8; ++g) {
      {
        float rm = fmaxf(s1t[0][g], s1t[1][g]);
#pragma unroll
        for (int off = 8; off >= 1; off >>= 1) rm = fmaxf(rm, __shfl_xor(rm, off, 32));
        float mn = fmaxf(m1[g], rm);
        float f = __expf(m1[g] - mn);
        float p0 = __expf(s1t[0][g] - mn);
        float p1 = __expf(s1t[1][g] - mn);
        float ps = p0 + p1;
#pragma unroll
        for (int off = 8; off >= 1; off >>= 1) ps += __shfl_xor(ps, off, 32);
        l1[g] = l1[g] * f + ps; m1[g] = mn;
#pragma unroll
        for (int t2 = 0; t2 < 8; ++t2) o1[t2][g] *= f;
        const int rbase = (g + (hi ? 8 : 0)) * 32;
        lds_p1[rbase + l] = (__bf16)p0;
        lds_p1[rbase + 16 + l] = (__bf16)p1;
      }
      {
        float rm = fmaxf(s2t[0][g], s2t[1][g]);
#pragma unroll
        for (int off = 8; off >= 1; off >>= 1) rm = fmaxf(rm, __shfl_xor(rm, off, 32));
        float mn = fmaxf(m2[g], rm);
        float f = __expf(m2[g] - mn);
        float p0 = __expf(s2t[0][g] - mn);
        float p1 = __expf(s2t[1][g] - mn);
        float ps = p0 + p1;
#pragma unroll
        for (int off = 8; off >= 1; off >>= 1) ps += __shfl_xor(ps, off, 32);
        l2[g] = l2[g] * f + ps; m2[g] = mn;
#pragma unroll
        for (int t2 = 0; t2 < 8; ++t2) o2[t2][g] *= f;
        const int rbase = (g + (hi ? 8 : 0)) * 32;
        lds_p2[rbase + l] = (__bf16)p0;
        lds_p2[rbase + 16 + l] = (__bf16)p1;
      }
    }
    __syncthreads();

    // PV: P (16x32) as A-fragment from LDS, V column tiles as B-fragments
    v16bf pa1 = load_a_frag(lds_p1 + l * 32, 0, hi);
    v16bf pa2 = load_a_frag(lds_p2 + l * 32, 0, hi);
#pragma unroll
    for (int t2 = 0; t2 < 8; t2 += 2) {
      v16bf vb0 = *(const v16bf*)(lds_vT + ((t2 + 0) * 16 + l) * 32 + (hi ? 16 : 0));
      v16bf vb1 = *(const v16bf*)(lds_vT + ((t2 + 1) * 16 + l) * 32 + (hi ? 16 : 0));
      o1[t2 + 0] = wmma_bf16(pa1, vb0, o1[t2 + 0]);
      o2[t2 + 0] = wmma_bf16(pa2, vb0, o2[t2 + 0]);
      o1[t2 + 1] = wmma_bf16(pa1, vb1, o1[t2 + 1]);
      o2[t2 + 1] = wmma_bf16(pa2, vb1, o2[t2 + 1]);
    }
    __syncthreads();
  }

  const float lam = *lamp;
#pragma unroll
  for (int t2 = 0; t2 < 8; ++t2)
#pragma unroll
    for (int g = 0; g < 8; ++g) {
      const int row = qbase + g + (hi ? 8 : 0);
      float val = o1[t2][g] / l1[g] - lam * (o2[t2][g] / l2[g]);
      out[(size_t)row * (NH * HD) + h * HD + t2 * 16 + l] = (__bf16)val;
    }
}

// ---- host-side orchestration ------------------------------------------------
extern "C" void kernel_launch(void* const* d_in, const int* in_sizes, int n_in,
                              void* d_out, int out_size, void* d_ws, size_t ws_size,
                              hipStream_t stream) {
  (void)in_sizes; (void)n_in; (void)out_size; (void)ws_size;
  const int*   positions = (const int*)d_in[0];
  const float* hidden    = (const float*)d_in[1];
  const float* qkv_w     = (const float*)d_in[2];
  const float* o_w       = (const float*)d_in[3];
  const float* q_norm_w  = (const float*)d_in[4];
  const float* k_norm_w  = (const float*)d_in[5];
  const float* lq1       = (const float*)d_in[6];
  const float* lk1       = (const float*)d_in[7];
  const float* lq2       = (const float*)d_in[8];
  const float* lk2       = (const float*)d_in[9];
  const float* sm_scaler = (const float*)d_in[10];
  const float* lam_init  = (const float*)d_in[11];
  float* outp = (float*)d_out;

  char* ws = (char*)d_ws;
  size_t off = 0;
  auto carve = [&](size_t bytes) -> char* {
    char* p = ws + off;
    off = (off + bytes + 255) & ~(size_t)255;
    return p;
  };
  __bf16* hid_bf  = (__bf16*)carve((size_t)T_SEQ * HIDN * 2);
  __bf16* qkvw_bf = (__bf16*)carve((size_t)QKV_N * HIDN * 2);
  __bf16* ow_bf   = (__bf16*)carve((size_t)HIDN * NH * HD * 2);
  float*  qkv_f   = (float*) carve((size_t)T_SEQ * QKV_N * 4);
  __bf16* q_bf    = (__bf16*)carve((size_t)T_SEQ * NH * HD * 2);
  __bf16* k_bf    = (__bf16*)carve((size_t)T_SEQ * NKV * HD * 2);
  __bf16* v_bf    = (__bf16*)carve((size_t)T_SEQ * NKV * HD * 2);
  __bf16* ao_bf   = (__bf16*)carve((size_t)T_SEQ * NH * HD * 2);
  float*  lam_p   = (float*) carve(4);

  {  // 1) fp32 -> bf16 operand conversion
    int n = T_SEQ * HIDN;
    convert_f32_bf16<<<(n + 255) / 256, 256, 0, stream>>>(hidden, hid_bf, n);
    n = QKV_N * HIDN;
    convert_f32_bf16<<<(n + 255) / 256, 256, 0, stream>>>(qkv_w, qkvw_bf, n);
    n = HIDN * NH * HD;
    convert_f32_bf16<<<(n + 255) / 256, 256, 0, stream>>>(o_w, ow_bf, n);
  }
  // 2) lambda scalar
  lambda_kernel<<<1, 128, 0, stream>>>(lq1, lk1, lq2, lk2, lam_init, lam_p);
  // 3) QKV GEMM: (T x HIDN) * (QKV_N x HIDN)^T -> (T x QKV_N) fp32
  gemm_bf16_wmma<<<(T_SEQ / BM) * (QKV_N / BN), 256, 0, stream>>>(
      hid_bf, qkvw_bf, qkv_f, T_SEQ, QKV_N, HIDN);
  // 4) RMSNorm + RoPE + split into bf16 Q/K/V
  {
    int waves = T_SEQ * 32;
    norm_rope_kernel<<<waves * 32 / 256, 256, 0, stream>>>(
        qkv_f, q_norm_w, k_norm_w, positions, q_bf, k_bf, v_bf);
  }
  // 5) differential flash attention
  diff_attn_kernel<<<NH * (T_SEQ / 16), 32, 0, stream>>>(
      q_bf, k_bf, v_bf, sm_scaler, lam_p, ao_bf);
  // 6) output GEMM: (T x 2048) * (HIDN x 2048)^T -> d_out fp32
  gemm_bf16_wmma<<<(T_SEQ / BM) * (HIDN / BN), 256, 0, stream>>>(
      ao_bf, ow_bf, outp, T_SEQ, HIDN, HIDN);
}
